// CrossAttention_20023137534338
// MI455X (gfx1250) — matrix operand order repaired and checked by reading, not compiled, not run on hardware
//
#include <hip/hip_runtime.h>

typedef __attribute__((ext_vector_type(16))) _Float16 v16h;
typedef __attribute__((ext_vector_type(8)))  float    v8f;
typedef __attribute__((ext_vector_type(4)))  float    v4f;
typedef _Float16 half_t;

namespace {

constexpr int Bn = 4, SQ = 2048, SKV = 2048, D = 512, H = 8, DH = 64;
constexpr int LDS_PAD = 4;
constexpr int LDS_ROW = SKV + LDS_PAD;  // row stride of score strip in LDS
constexpr float EPS   = 1e-5f;
constexpr float SCALE = 0.125f;  // DH^-0.5

// ---- WMMA fragment helpers (wave32, v_wmma_f32_16x16x32_f16) -----------

// A-matrix 16x32 f16 half-pair K map (ISA 7.12.2): lanes 0-15 / 16-31 split.
__device__ __forceinline__ int kmap(int j, int lane) {
  int khi = (lane & 16) ? 8 : 0;
  return (j < 4) ? (khi + 2 * j) : (16 + khi + 2 * (j - 4));
}

// A fragment from row-major MxK f16 source.
__device__ __forceinline__ v16h load_a_h(const half_t* __restrict__ A, int m0,
                                         int k0, int lda, int lane) {
  const half_t* base = A + (size_t)(m0 + (lane & 15)) * lda + k0;
  v16h a;
#pragma unroll
  for (int j = 0; j < 8; ++j) {
    int k = kmap(j, lane);
    a[2 * j]     = base[k];
    a[2 * j + 1] = base[k + 1];
  }
  return a;
}

// A fragment from row-major MxK f32 source (convert on load; works for LDS).
__device__ __forceinline__ v16h load_a_f32(const float* A, int m0, int k0,
                                           int lda, int lane) {
  const float* base = A + (size_t)(m0 + (lane & 15)) * lda + k0;
  v16h a;
#pragma unroll
  for (int j = 0; j < 8; ++j) {
    int k = kmap(j, lane);
    a[2 * j]     = (half_t)base[k];
    a[2 * j + 1] = (half_t)base[k + 1];
  }
  return a;
}

// B fragment 32x16: column n = lane % 16, K over the 16 halves in the same
// half-pair order as the A fragment (strided loads from row-major KxN source).
__device__ __forceinline__ v16h load_b_kn(const half_t* __restrict__ Bm, int k0,
                                          int n0, int ldb, int lane) {
  const half_t* base = Bm + (size_t)k0 * ldb + n0 + (lane & 15);
  int khi = (lane & 16) ? 8 : 0;
  v16h b;
#pragma unroll
  for (int j = 0; j < 16; ++j) {
    int k = (j < 8) ? (khi + j) : (16 + khi + (j - 8));
    b[j] = base[(size_t)k * ldb];
  }
  return b;
}

__device__ __forceinline__ v8f wmma_f16(v16h a, v16h b, v8f c) {
  return __builtin_amdgcn_wmma_f32_16x16x32_f16(false, a, false, b, (short)0, c,
                                                false, false);
}

// ---- Kernels ------------------------------------------------------------

// LayerNorm over D=512, one block (256 threads) per row, f16 output.
__global__ void ln_to_h(const float* __restrict__ x, const float* __restrict__ g,
                        const float* __restrict__ bb, half_t* __restrict__ out) {
  int row = blockIdx.x, tid = threadIdx.x;
  const float* xr = x + (size_t)row * D;
  float v0 = xr[tid], v1 = xr[tid + 256];
  __shared__ float sh[256];
  sh[tid] = v0 + v1;
  __syncthreads();
  for (int off = 128; off > 0; off >>= 1) {
    if (tid < off) sh[tid] += sh[tid + off];
    __syncthreads();
  }
  float mean = sh[0] * (1.0f / D);
  __syncthreads();
  float d0 = v0 - mean, d1 = v1 - mean;
  sh[tid] = d0 * d0 + d1 * d1;
  __syncthreads();
  for (int off = 128; off > 0; off >>= 1) {
    if (tid < off) sh[tid] += sh[tid + off];
    __syncthreads();
  }
  float rstd = rsqrtf(sh[0] * (1.0f / D) + EPS);
  half_t* orow = out + (size_t)row * D;
  orow[tid]       = (half_t)(d0 * rstd * g[tid] + bb[tid]);
  orow[tid + 256] = (half_t)(d1 * rstd * g[tid + 256] + bb[tid + 256]);
}

__global__ void cvt_f32_h(const float* __restrict__ in, half_t* __restrict__ out,
                          int n) {
  int i = blockIdx.x * 256 + threadIdx.x;
  if (i < n) out[i] = (half_t)in[i];
}

// C_h(MxN) = A_h(MxK) @ W_h(KxN), K=N=D=512. Block: 4 waves x 16x16 tiles.
__global__ void gemm_qkv(const half_t* __restrict__ A,
                         const half_t* __restrict__ W,
                         half_t* __restrict__ C) {
  int lane = threadIdx.x & 31, wave = threadIdx.x >> 5;
  int n0 = blockIdx.x * 64 + wave * 16;
  int m0 = blockIdx.y * 16;
  v8f c = {};
  for (int k0 = 0; k0 < D; k0 += 32) {
    v16h a = load_a_h(A, m0, k0, D, lane);
    v16h b = load_b_kn(W, k0, n0, D, lane);
    c = wmma_f16(a, b, c);
  }
  int n = n0 + (lane & 15), mb = m0 + ((lane & 16) ? 8 : 0);
#pragma unroll
  for (int r = 0; r < 8; ++r) C[(size_t)(mb + r) * D + n] = (half_t)c[r];
}

// K (b,kv,h,d) -> Kt (bh, d, kv): makes QK^T B-fragments contiguous loads.
__global__ void transpose_k(const half_t* __restrict__ Kh,
                            half_t* __restrict__ Kt) {
  size_t idx = (size_t)blockIdx.x * 256 + threadIdx.x;  // Kt linear index
  int kv = idx & (SKV - 1);
  size_t t = idx >> 11;          // bh*DH + d
  int d  = (int)(t & (DH - 1));
  int bh = (int)(t >> 6);
  int b = bh >> 3, h = bh & 7;
  Kt[idx] = Kh[(size_t)(b * SKV + kv) * D + h * DH + d];
}

// Fused flash-style: scores (WMMA) -> softmax (LDS) -> NT attn write (once)
// -> P@V (WMMA from LDS, unnormalized; normalize at writeout).
// One block = 16 q-rows x full SKV for one (b,h).
__global__ void attn_fused(const half_t* __restrict__ Q,
                           const half_t* __restrict__ Kt,
                           const half_t* __restrict__ V,
                           float* __restrict__ attn,
                           half_t* __restrict__ ctx) {
  extern __shared__ float S[];          // [16][LDS_ROW] score strip (~128 KB)
  __shared__ float red[256];
  __shared__ float rowstat[16];
  __shared__ float red2[4 * 32 * 8];    // split-K partial accumulators

  int tid = threadIdx.x;
  int lane = tid & 31, wave = tid >> 5;
  int q0 = blockIdx.x * 16;
  int bh = blockIdx.y, b = bh >> 3, h = bh & 7;

  const half_t* Qb  = Q  + (size_t)b * SQ * D + h * DH;   // [q, d]  ld = D
  const half_t* KtB = Kt + (size_t)bh * DH * SKV;         // [d, kv] ld = SKV
  const half_t* Vb  = V  + (size_t)b * SKV * D + h * DH;  // [kv, d] ld = D

  // ---- Phase 1: S = scale * Q K^T (8 waves x 16 kv-tiles each) ----
  v16h aq0 = load_a_h(Qb, q0, 0, D, lane);
  v16h aq1 = load_a_h(Qb, q0, 32, D, lane);
  int nloc = lane & 15, mloc = (lane & 16) ? 8 : 0;
  for (int t = 0; t < 16; ++t) {
    int kv0 = (wave * 16 + t) * 16;
    v8f c = {};
    c = wmma_f16(aq0, load_b_kn(KtB, 0, kv0, SKV, lane), c);
    c = wmma_f16(aq1, load_b_kn(KtB, 32, kv0, SKV, lane), c);
#pragma unroll
    for (int r = 0; r < 8; ++r)
      S[(mloc + r) * LDS_ROW + kv0 + nloc] = c[r] * SCALE;
  }
  __syncthreads();

  // ---- Phase 2: softmax stats per row; S keeps unnormalized exp ----
  int row = tid >> 4, sub = tid & 15;
  float m = -3.4e38f;
  for (int i = sub; i < SKV; i += 16)
    m = fmaxf(m, S[row * LDS_ROW + i]);
  red[tid] = m;
  __syncthreads();
  if (sub == 0) {
    float mm = red[row * 16];
    for (int j = 1; j < 16; ++j) mm = fmaxf(mm, red[row * 16 + j]);
    rowstat[row] = mm;
  }
  __syncthreads();
  m = rowstat[row];
  float s = 0.f;
  for (int i = sub; i < SKV; i += 16) {
    float e = __expf(S[row * LDS_ROW + i] - m);
    S[row * LDS_ROW + i] = e;
    s += e;
  }
  red[tid] = s;
  __syncthreads();
  if (sub == 0) {
    float ss = 0.f;
    for (int j = 0; j < 16; ++j) ss += red[row * 16 + j];
    rowstat[row] = 1.0f / ss;
  }
  __syncthreads();

  // Single coalesced non-temporal float4 write of normalized attn to HBM.
  // (attn is write-once/stream-out: NT keeps it from evicting L2-resident
  //  Q/K/V/ctx working set.)
  float* attn_g = attn + ((size_t)bh * SQ + q0) * SKV;
  for (int i4 = tid; i4 < 16 * (SKV / 4); i4 += 256) {
    int r2 = i4 >> 9, c4 = (i4 & 511) * 4;
    float inv = rowstat[r2];
    const float* sp = &S[r2 * LDS_ROW + c4];
    v4f pv = {sp[0] * inv, sp[1] * inv, sp[2] * inv, sp[3] * inv};
    __builtin_nontemporal_store(pv, (v4f*)(attn_g + (size_t)r2 * SKV + c4));
  }
  __syncthreads();

  // ---- Phase 3: ctx = P @ V with unnormalized P; split-K over 8 waves ----
  int wv = wave & 3, halfk = wave >> 2;
  int n0 = wv * 16;
  int kbeg = halfk * (SKV / 2);
  v8f c = {};
  for (int k0 = kbeg; k0 < kbeg + SKV / 2; k0 += 32) {
    v16h a  = load_a_f32(S, 0, k0, LDS_ROW, lane);  // exp(S) from LDS, ->f16
    v16h bf = load_b_kn(Vb, k0, n0, D, lane);
    c = wmma_f16(a, bf, c);
  }
  if (halfk == 1) {
#pragma unroll
    for (int r = 0; r < 8; ++r) red2[(wv * 32 + lane) * 8 + r] = c[r];
  }
  __syncthreads();
  if (halfk == 0) {
    half_t* Cb = ctx + (size_t)b * SQ * D + h * DH;
    int n = n0 + nloc, mb = q0 + mloc;
#pragma unroll
    for (int r = 0; r < 8; ++r) {
      float v = (c[r] + red2[(wv * 32 + lane) * 8 + r]) * rowstat[mloc + r];
      Cb[(size_t)(mb + r) * D + n] = (half_t)v;
    }
  }
}

// out = ctx @ Wo + bo  (f32 out)
__global__ void out_proj(const half_t* __restrict__ ctx,
                         const half_t* __restrict__ Wo,
                         const float* __restrict__ bo, float* __restrict__ out) {
  int lane = threadIdx.x & 31, wave = threadIdx.x >> 5;
  int n0 = blockIdx.x * 64 + wave * 16;
  int m0 = blockIdx.y * 16;
  v8f c = {};
  for (int k0 = 0; k0 < D; k0 += 32) {
    v16h a = load_a_h(ctx, m0, k0, D, lane);
    v16h b = load_b_kn(Wo, k0, n0, D, lane);
    c = wmma_f16(a, b, c);
  }
  int n = n0 + (lane & 15), mb = m0 + ((lane & 16) ? 8 : 0);
  float bias = bo[n];
#pragma unroll
  for (int r = 0; r < 8; ++r) out[(size_t)(mb + r) * D + n] = c[r] + bias;
}

}  // namespace

extern "C" void kernel_launch(void* const* d_in, const int* in_sizes, int n_in,
                              void* d_out, int out_size, void* d_ws,
                              size_t ws_size, hipStream_t stream) {
  (void)in_sizes; (void)n_in; (void)out_size; (void)ws_size;
  const float* x       = (const float*)d_in[0];
  const float* y       = (const float*)d_in[1];
  const float* ln_q_g  = (const float*)d_in[2];
  const float* ln_q_b  = (const float*)d_in[3];
  const float* ln_kv_g = (const float*)d_in[4];
  const float* ln_kv_b = (const float*)d_in[5];
  const float* Wq      = (const float*)d_in[6];
  const float* Wk      = (const float*)d_in[7];
  const float* Wv      = (const float*)d_in[8];
  const float* Wo      = (const float*)d_in[9];
  const float* bo      = (const float*)d_in[10];

  float* out  = (float*)d_out;
  float* attn = out + (size_t)Bn * SQ * D;  // tuple order: (out, attn)

  half_t* p    = (half_t*)d_ws;
  half_t* xn   = p; p += (size_t)Bn * SQ  * D;
  half_t* yn   = p; p += (size_t)Bn * SKV * D;
  half_t* Wq_h = p; p += (size_t)D * D;
  half_t* Wk_h = p; p += (size_t)D * D;
  half_t* Wv_h = p; p += (size_t)D * D;
  half_t* Wo_h = p; p += (size_t)D * D;
  half_t* Qh   = p; p += (size_t)Bn * SQ  * D;
  half_t* Kh   = p; p += (size_t)Bn * SKV * D;
  half_t* Vh   = p; p += (size_t)Bn * SKV * D;
  half_t* Kt   = p; p += (size_t)Bn * H * DH * SKV;
  half_t* ctx  = p; p += (size_t)Bn * SQ  * D;

  // 1) LayerNorm (+f16 convert)
  ln_to_h<<<Bn * SQ, 256, 0, stream>>>(x, ln_q_g, ln_q_b, xn);
  ln_to_h<<<Bn * SKV, 256, 0, stream>>>(y, ln_kv_g, ln_kv_b, yn);

  // 2) Weight conversion
  int wn = D * D;
  cvt_f32_h<<<(wn + 255) / 256, 256, 0, stream>>>(Wq, Wq_h, wn);
  cvt_f32_h<<<(wn + 255) / 256, 256, 0, stream>>>(Wk, Wk_h, wn);
  cvt_f32_h<<<(wn + 255) / 256, 256, 0, stream>>>(Wv, Wv_h, wn);
  cvt_f32_h<<<(wn + 255) / 256, 256, 0, stream>>>(Wo, Wo_h, wn);

  // 3) Q/K/V projections (WMMA)
  gemm_qkv<<<dim3(D / 64, (Bn * SQ) / 16), 128, 0, stream>>>(xn, Wq_h, Qh);
  gemm_qkv<<<dim3(D / 64, (Bn * SKV) / 16), 128, 0, stream>>>(yn, Wk_h, Kh);
  gemm_qkv<<<dim3(D / 64, (Bn * SKV) / 16), 128, 0, stream>>>(yn, Wv_h, Vh);

  // 4) K transpose into [bh][d][kv]
  transpose_k<<<(Bn * H * DH * SKV) / 256, 256, 0, stream>>>(Kh, Kt);

  // 5) fused scores + softmax + attn-write + P@V
  size_t smem = (size_t)16 * LDS_ROW * sizeof(float);  // ~128 KB
  attn_fused<<<dim3(SQ / 16, Bn * H), 256, smem, stream>>>(Qh, Kt, Vh, attn,
                                                           ctx);

  // 6) out = ctx @ Wo + bo (WMMA)
  out_proj<<<dim3(D / 64, (Bn * SQ) / 16), 128, 0, stream>>>(ctx, Wo_h, bo, out);
}
